// CorrelationModule_31842887533353
// MI455X (gfx1250) — matrix operand order repaired and checked by reading, not compiled, not run on hardware
//
#include <hip/hip_runtime.h>

// CDNA5 / gfx1250 correlation layer via banded f16 WMMA GEMM.
// out[b, du*21+dv, i, j] = sum_c x1[b,c,i,j] * x2[b,c,i+du-10,j+dv-10] (zero pad)
//
// Mapping: per (b, i, du, 16-wide j-tile) compute G[16x48] = A[16x128] * B[128x48]
// with A = x1 tile (rows = j), B = 48-col slab of the shifted x2 row, then
// extract the Hankel band out[jj,dv] = G[jj, jj+dv].  f16 inputs / f32 accumulate
// (inputs ~N(0,1): no overflow, ~1e-3 rel err) -> v_wmma_f32_16x16x32_f16.
//
// Zero padding is done BRANCHLESSLY: addresses are clamped into range so loads
// are unconditional (coalesced, batched, one wait per fragment) and the value
// is zeroed with a 0/1 fp32 mask multiply before the f16 convert.

typedef __attribute__((ext_vector_type(16))) _Float16 v16h;
typedef __attribute__((ext_vector_type(8)))  float    v8f;

#define CORR_WIN 21
#define CORR_PAD 10
#define NC 128   // channels
#define NW 128   // height (i)
#define NH 128   // width  (j)
#define NB 4     // batch

__global__ __launch_bounds__(64, 1)
void CorrelationModule_31842887533353_kernel(const float* __restrict__ x1,
                                             const float* __restrict__ x2,
                                             float* __restrict__ out) {
    const int jt = blockIdx.x;        // j-tile: 0..7  (16 cols each)
    const int i  = blockIdx.y;        // 0..127
    const int b  = blockIdx.z;        // 0..3
    const int j0 = jt << 4;

    const int lane = threadIdx.x & 31;
    const int wav  = threadIdx.x >> 5;   // du parity (2 waves / block)
    const int lo   = lane & 15;
    const int hi   = lane >> 4;

    const size_t chs = (size_t)NW * NH;            // channel stride (16384)
    const float* x1p = x1 + (size_t)b * NC * chs + (size_t)i * NH + (j0 + lo);
    const float* x2b = x2 + (size_t)b * NC * chs;

    // ---- A = x1[b, :, i, j0:j0+16]^T as four 16x32 f16 fragments -----------
    // ISA 16-bit A 16x32 layout: lane holds row M=lane%16;
    //   lanes 0-15:  elems 0..7 -> K=0..7,  elems 8..15 -> K=16..23
    //   lanes 16-31: elems 0..7 -> K=8..15, elems 8..15 -> K=24..31
    v16h afrag[4];
    #pragma unroll
    for (int kb = 0; kb < 4; ++kb) {
        float tmp[16];
        #pragma unroll
        for (int e = 0; e < 16; ++e) {
            const int k = ((e >> 3) << 4) + (hi << 3) + (e & 7);
            tmp[e] = x1p[(size_t)(kb * 32 + k) * chs];
        }
        #pragma unroll
        for (int e = 0; e < 16; ++e) afrag[kb][e] = (_Float16)tmp[e];
    }

    // one du per wave-iteration: wave 0 -> 0,2,..,20 ; wave 1 -> 1,3,..,19
    for (int du = wav; du < CORR_WIN; du += 2) {
        const int row    = i + du - CORR_PAD;
        const int rowc   = row < 0 ? 0 : (row > NW - 1 ? NW - 1 : row);
        const bool rowok = (row >= 0) && (row < NW);
        const float* x2r = x2b + (size_t)rowc * NH;

        // ---- G[16 x 48] = A[16 x 128] * B[128 x 48] ------------------------
        v8f acc[3];
        #pragma unroll
        for (int t = 0; t < 3; ++t) {
            v8f a = {0.f, 0.f, 0.f, 0.f, 0.f, 0.f, 0.f, 0.f};
            const int   q    = j0 - CORR_PAD + t * 16 + lo;   // x2 column
            const int   qc   = q < 0 ? 0 : (q > NH - 1 ? NH - 1 : q);
            const float msk  = (rowok && q >= 0 && q < NH) ? 1.0f : 0.0f;
            const float* x2q = x2r + qc;
            #pragma unroll
            for (int kb = 0; kb < 4; ++kb) {
                // ISA 16-bit B 32x16 layout: lane n holds col N=n%16;
                // lanes 0-15: elems 0..7 -> K=0..7, 8..15 -> K=16..23; lanes 16-31: K=8..15, 24..31
                float tmp[16];
                #pragma unroll
                for (int e = 0; e < 16; ++e) {
                    const int c = kb * 32 + ((e >> 3) << 4) + (hi << 3) + (e & 7);
                    tmp[e] = x2q[(size_t)c * chs];   // unconditional, coalesced
                }
                v16h bfrag;
                #pragma unroll
                for (int e = 0; e < 16; ++e)
                    bfrag[e] = (_Float16)(tmp[e] * msk);  // branchless zero-pad
                // D = A*B + C  (f32 accumulate)
                a = __builtin_amdgcn_wmma_f32_16x16x32_f16(
                        /*neg_a=*/false, afrag[kb],
                        /*neg_b=*/false, bfrag,
                        /*c_mod=*/(short)0, a,
                        /*reuse_a=*/false, /*reuse_b=*/false);
            }
            acc[t] = a;
        }

        // ---- band extraction: out[jj, dv] = G[jj, jj + dv] -----------------
        // C fragment: elem r of lane -> M = r + 8*hi, N = 16*t + lo
        float* outb = out + (size_t)b * (CORR_WIN * CORR_WIN) * chs
                          + (size_t)i * NH + j0;
        #pragma unroll
        for (int t = 0; t < 3; ++t) {
            #pragma unroll
            for (int r = 0; r < 8; ++r) {
                const int m  = r + (hi << 3);          // jj within tile
                const int dv = 16 * t + lo - m;        // displacement
                if (dv >= 0 && dv < CORR_WIN) {
                    const int uv = du * CORR_WIN + dv;
                    outb[(size_t)uv * chs + m] = acc[t][r];
                }
            }
        }
    }
}

extern "C" void kernel_launch(void* const* d_in, const int* in_sizes, int n_in,
                              void* d_out, int out_size, void* d_ws, size_t ws_size,
                              hipStream_t stream) {
    (void)in_sizes; (void)n_in; (void)d_ws; (void)ws_size; (void)out_size;
    const float* x1 = (const float*)d_in[0];
    const float* x2 = (const float*)d_in[1];
    float* out = (float*)d_out;

    dim3 grid(NH / 16, NW, NB);   // (j-tiles, i, b) = (8, 128, 4)
    dim3 block(64);               // 2 waves: du parity split, A reused across 21 du
    CorrelationModule_31842887533353_kernel<<<grid, block, 0, stream>>>(x1, x2, out);
}
